// MaskedPiecewiseQuadraticAutoregressiveTransform_91319594647697
// MI455X (gfx1250) — compile-verified, matrix operand order checked by reading, not run on hardware
//
#include <hip/hip_runtime.h>
#include <hip/hip_bf16.h>

typedef __attribute__((ext_vector_type(16))) __bf16 v16bf;
typedef __attribute__((ext_vector_type(8)))  float  v8f;

#define FEATS   256
#define HID     256
#define NBINS   10
#define MULTP   21
#define BATCHN  8192
#define KT      8          // 256 / 32 k-tiles
#define APAD    264        // padded row stride (halves) for LDS activation tiles
#define TILE_H  512        // halves per 16x32 bf16 fragment tile
#define SCALE_P 0.0625f    // 1/sqrt(256)

union BF16x16 { v16bf v; unsigned int u[8]; unsigned short h[16]; };

__device__ __forceinline__ unsigned short f2bf(float x) {
    union { float f; unsigned int u; } t; t.f = x;
    unsigned int r = t.u + 0x7FFFu + ((t.u >> 16) & 1u);
    return (unsigned short)(r >> 16);
}

// K offset within a 32-wide k-tile for (vgpr v, half) per the CDNA5 16-bit A/B layout
__device__ __forceinline__ int frag_k0(int v, int half) {
    return ((v < 4) ? 0 : 16) + ((v & 3) * 2) + 8 * half;
}

// Gather an A fragment (16x32 bf16) from a row-major padded LDS tile
__device__ __forceinline__ v16bf load_a_frag(const unsigned short* lds, int kt, int lane) {
    const int m = lane & 15, half = lane >> 4;
    BF16x16 r;
#pragma unroll
    for (int v = 0; v < 8; ++v) {
        const int k0 = frag_k0(v, half);
        r.u[v] = *(const unsigned int*)(lds + m * APAD + kt * 32 + k0);
    }
    return r.v;
}

__device__ __forceinline__ v16bf load_b_frag(const unsigned short* tileBase, int lane) {
    return *(const v16bf*)(tileBase + lane * 16);
}

// ---------------------------------------------------------------------------
// Weight prep: mask + bf16 + swizzle into fragment order
// ---------------------------------------------------------------------------
__global__ void prep_square(const float* __restrict__ W0, const float* __restrict__ Wb,
                            unsigned short* __restrict__ w0f, unsigned short* __restrict__ wbf) {
    const long idx = (long)blockIdx.x * 256 + threadIdx.x;     // 5*16*8*512 total
    const int  e     = (int)(idx & 511);
    const long tile  = idx >> 9;                               // 0..639
    const int  ktile = (int)(tile & 7);
    const int  otile = (int)((tile >> 3) & 15);
    const int  mat   = (int)(tile >> 7);                       // 0..4
    const int  lane  = e >> 4, slot = e & 15;
    const int  half  = lane >> 4, n = lane & 15;
    const int  v     = slot >> 1, pbit = slot & 1;
    const int  k     = ktile * 32 + frag_k0(v, half) + pbit;
    const int  o     = otile * 16 + n;
    const int  hd_o  = (o % 255) + 1;
    const int  hd_k  = (k % 255) + 1;
    float wv; int msk;
    if (mat == 0) { wv = W0[o * 256 + k];                    msk = (hd_o >= (k + 1)); }
    else          { wv = Wb[(mat - 1) * 65536 + o * 256 + k]; msk = (hd_o >= hd_k);   }
    unsigned short* dst = (mat == 0) ? w0f : (wbf + (long)(mat - 1) * 65536);
    dst[(long)(otile * KT + ktile) * TILE_H + e] = msk ? f2bf(wv) : (unsigned short)0;
}

__global__ void prep_wf(const float* __restrict__ Wf, unsigned short* __restrict__ wff) {
    const long idx = (long)blockIdx.x * 256 + threadIdx.x;     // 256*8*2*512 total
    const int  e     = (int)(idx & 511);
    const long tile  = idx >> 9;                               // 0..4095
    const int  nt    = (int)(tile & 1);
    const int  ktile = (int)((tile >> 1) & 7);
    const int  f     = (int)(tile >> 4);
    const int  lane  = e >> 4, slot = e & 15;
    const int  half  = lane >> 4, nn = lane & 15;
    const int  v     = slot >> 1, pbit = slot & 1;
    const int  k     = ktile * 32 + frag_k0(v, half) + pbit;
    const int  Np    = nt * 16 + nn;                           // padded output within feature (0..31)
    unsigned short r = 0;
    if (Np < MULTP) {
        const int o = f * MULTP + Np;
        const int outdeg = f + 1;
        const int hd = (k % 255) + 1;
        if (outdeg > hd) r = f2bf(Wf[(long)o * 256 + k]);
    }
    wff[(long)(f * 16 + ktile * 2 + nt) * TILE_H + e] = r;
}

// ---------------------------------------------------------------------------
// Trunk: one [16,256]x[256,256] GEMM stage (wave owns 64 output columns)
// mode 0: hF = val+b, relu->outB | 1: relu(val+b)->outB
// mode 2: hF += val+b, relu->outB | 3: hF+val+b -> bf16 global hbf
// ---------------------------------------------------------------------------
__device__ __forceinline__ void gemm256(const unsigned short* aLds,
                                        const unsigned short* __restrict__ frags,
                                        const float* __restrict__ bias,
                                        float* hF, unsigned short* outB,
                                        unsigned short* __restrict__ hbfG,
                                        int rowBase, int mode, int wave, int lane) {
    const int n = lane & 15, half = lane >> 4;
#pragma unroll
    for (int t = 0; t < 4; ++t) {
        const int ot = wave * 4 + t;
        v8f acc = {};
#pragma unroll
        for (int kt = 0; kt < KT; ++kt) {
            v16bf a = load_a_frag(aLds, kt, lane);
            v16bf b = load_b_frag(frags + (long)(ot * KT + kt) * TILE_H, lane);
            acc = __builtin_amdgcn_wmma_f32_16x16x32_bf16(false, a, false, b,
                                                          (short)0, acc, false, false);
        }
        const int ncol = ot * 16 + n;
        const float bv = bias[ncol];
#pragma unroll
        for (int j = 0; j < 8; ++j) {
            const int m = j + 8 * half;
            float val = acc[j] + bv;
            if (mode == 0) {
                hF[m * HID + ncol] = val;
                outB[m * APAD + ncol] = f2bf(fmaxf(val, 0.f));
            } else if (mode == 1) {
                outB[m * APAD + ncol] = f2bf(fmaxf(val, 0.f));
            } else {
                float h = hF[m * HID + ncol] + val;
                if (mode == 2) {
                    hF[m * HID + ncol] = h;
                    outB[m * APAD + ncol] = f2bf(fmaxf(h, 0.f));
                } else {
                    hbfG[(long)(rowBase + m) * HID + ncol] = f2bf(h);
                }
            }
        }
    }
}

__global__ void __launch_bounds__(128)
trunk_kernel(const float* __restrict__ x, const float* __restrict__ b0,
             const float* __restrict__ bb,
             const unsigned short* __restrict__ w0f,
             const unsigned short* __restrict__ wbf,
             unsigned short* __restrict__ hbf) {
    __shared__ __align__(16) unsigned short bufA[16 * APAD];
    __shared__ __align__(16) unsigned short bufB[16 * APAD];
    __shared__ __align__(16) float hF[16 * HID];

    const int tid = threadIdx.x, wave = tid >> 5, lane = tid & 31;
    const int rowBase = blockIdx.x * 16;

    for (int i = tid; i < 16 * FEATS; i += 128) {
        const int r = i >> 8, c = i & 255;
        bufA[r * APAD + c] = f2bf(x[(long)(rowBase + r) * FEATS + c]);
    }
    __syncthreads();

    gemm256(bufA, w0f,          b0,       hF, bufB, nullptr, rowBase, 0, wave, lane);
    __syncthreads();
    gemm256(bufB, wbf + 0*65536, bb + 0,   hF, bufA, nullptr, rowBase, 1, wave, lane);
    __syncthreads();
    gemm256(bufA, wbf + 1*65536, bb + 256, hF, bufB, nullptr, rowBase, 2, wave, lane);
    __syncthreads();
    gemm256(bufB, wbf + 2*65536, bb + 512, hF, bufA, nullptr, rowBase, 1, wave, lane);
    __syncthreads();
    gemm256(bufA, wbf + 3*65536, bb + 768, hF, nullptr, hbf,  rowBase, 3, wave, lane);
}

// ---------------------------------------------------------------------------
// Final projection (256 -> 21 per feature) fused with quadratic spline
// ---------------------------------------------------------------------------
__global__ void __launch_bounds__(128)
final_spline_kernel(const float* __restrict__ x, const unsigned short* __restrict__ hbf,
                    const unsigned short* __restrict__ wff, const float* __restrict__ bfb,
                    float* __restrict__ out, float* __restrict__ logdet) {
    __shared__ __align__(16) unsigned short hA[16 * APAD];
    __shared__ __align__(16) float xs[16 * FEATS];
    __shared__ __align__(16) float ysc[4][16 * 32];
    __shared__ float ldp[4][16];

    const int tid = threadIdx.x, wave = tid >> 5, lane = tid & 31;
    const int rowBase = blockIdx.x * 16;

    for (int i = tid; i < 16 * FEATS; i += 128) {
        const int r = i >> 8, c = i & 255;
        hA[r * APAD + c] = hbf[(long)(rowBase + r) * FEATS + c];
        xs[r * FEATS + c] = x[(long)(rowBase + r) * FEATS + c];
    }
    __syncthreads();

    // preload all A fragments of the h tile (reused across all 64 features)
    v16bf afr[KT];
#pragma unroll
    for (int kt = 0; kt < KT; ++kt) afr[kt] = load_a_frag(hA, kt, lane);

    const int n = lane & 15, half = lane >> 4;
    float ldacc = 0.f;

    for (int f = wave; f < FEATS; f += 4) {
        const unsigned short* fb = wff + (long)f * (KT * 2 * TILE_H);
        v8f acc0 = {}, acc1 = {};
#pragma unroll
        for (int kt = 0; kt < KT; ++kt) {
            v16bf bA = load_b_frag(fb + (long)(kt * 2 + 0) * TILE_H, lane);
            v16bf bB = load_b_frag(fb + (long)(kt * 2 + 1) * TILE_H, lane);
            acc0 = __builtin_amdgcn_wmma_f32_16x16x32_bf16(false, afr[kt], false, bA,
                                                           (short)0, acc0, false, false);
            acc1 = __builtin_amdgcn_wmma_f32_16x16x32_bf16(false, afr[kt], false, bB,
                                                           (short)0, acc1, false, false);
        }
#pragma unroll
        for (int j = 0; j < 8; ++j) {
            const int m = j + 8 * half;
            ysc[wave][m * 32 + n]      = acc0[j];
            ysc[wave][m * 32 + 16 + n] = acc1[j];
        }
        if (lane < 16) {   // 16 rows: one lane per row does the spline
            const int m = lane;
            const float xv = xs[m * FEATS + f];
            float p[MULTP];
#pragma unroll
            for (int j = 0; j < MULTP; ++j)
                p[j] = (ysc[wave][m * 32 + j] + bfb[f * MULTP + j]) * SCALE_P;

            // widths: softmax over p[0..9]
            float mx = p[0];
#pragma unroll
            for (int j = 1; j < NBINS; ++j) mx = fmaxf(mx, p[j]);
            float w[NBINS], s = 0.f;
#pragma unroll
            for (int j = 0; j < NBINS; ++j) { w[j] = __expf(p[j] - mx); s += w[j]; }
            const float invs = 1.f / s;
#pragma unroll
            for (int j = 0; j < NBINS; ++j)
                w[j] = 1e-3f + (1.f - 1e-3f * NBINS) * (w[j] * invs);

            // heights
            float he[NBINS + 1];
#pragma unroll
            for (int j = 0; j <= NBINS; ++j) he[j] = __expf(p[NBINS + j]);
            float area = 0.f;
#pragma unroll
            for (int j = 0; j < NBINS; ++j) area += 0.5f * (he[j] + he[j + 1]) * w[j];
            const float ia = 1.f / area;
            float hh[NBINS + 1];
#pragma unroll
            for (int j = 0; j <= NBINS; ++j) hh[j] = 1e-3f + (1.f - 1e-3f) * he[j] * ia;

            // cdf / knot locations with clamped tails
            float cdf[NBINS + 1], locs[NBINS + 1];
            cdf[0] = 0.f; locs[0] = 0.f;
            float cc = 0.f, ll = 0.f;
#pragma unroll
            for (int j = 0; j < NBINS; ++j) {
                cc += 0.5f * (hh[j] + hh[j + 1]) * w[j];
                ll += w[j];
                cdf[j + 1] = cc; locs[j + 1] = ll;
            }
            cdf[NBINS] = 1.f; locs[NBINS] = 1.f;

            int cnt = 0;
#pragma unroll
            for (int j = 0; j <= NBINS; ++j) cnt += (xv >= locs[j]) ? 1 : 0;
            int idx = cnt - 1;
            idx = idx < 0 ? 0 : (idx > NBINS - 1 ? NBINS - 1 : idx);

            const float wi = w[idx], hl = hh[idx], hr = hh[idx + 1];
            const float aq = 0.5f * (hr - hl) * wi;
            const float bq = hl * wi;
            const float al = (xv - locs[idx]) / wi;
            float o = aq * al * al + bq * al + cdf[idx];
            o = fminf(fmaxf(o, 0.f), 1.f);
            out[(long)(rowBase + m) * FEATS + f] = o;
            ldacc += __logf(al * (hr - hl) + hl);
        }
    }

    if (lane < 16) ldp[wave][lane] = ldacc;
    __syncthreads();
    if (tid < 16)
        logdet[rowBase + tid] = ldp[0][tid] + ldp[1][tid] + ldp[2][tid] + ldp[3][tid];
}

// ---------------------------------------------------------------------------
extern "C" void kernel_launch(void* const* d_in, const int* in_sizes, int n_in,
                              void* d_out, int out_size, void* d_ws, size_t ws_size,
                              hipStream_t stream) {
    (void)in_sizes; (void)n_in; (void)out_size; (void)ws_size;
    const float* x  = (const float*)d_in[0];
    const float* W0 = (const float*)d_in[1];
    const float* b0 = (const float*)d_in[2];
    const float* Wb = (const float*)d_in[3];
    const float* bb = (const float*)d_in[4];
    const float* Wf = (const float*)d_in[5];
    const float* bf = (const float*)d_in[6];

    unsigned short* ws  = (unsigned short*)d_ws;
    unsigned short* w0f = ws;                       // 16*8*512           = 65536 halves
    unsigned short* wbf = w0f + 65536;              // 4 * 65536          = 262144 halves
    unsigned short* wff = wbf + 262144;             // 256*16*512         = 2097152 halves
    unsigned short* hbf = wff + 2097152;            // 8192*256           = 2097152 halves

    float* out = (float*)d_out;
    float* ld  = out + (long)BATCHN * FEATS;

    prep_square<<<1280, 256, 0, stream>>>(W0, Wb, w0f, wbf);
    prep_wf<<<8192, 256, 0, stream>>>(Wf, wff);
    trunk_kernel<<<BATCHN / 16, 128, 0, stream>>>(x, b0, bb, w0f, wbf, hbf);
    final_spline_kernel<<<BATCHN / 16, 128, 0, stream>>>(x, hbf, wff, bf, out, ld);
}
